// EGNNLayer_8297876816265
// MI455X (gfx1250) — matrix operand order repaired and checked by reading, not compiled, not run on hardware
//
#include <hip/hip_runtime.h>
#include <hip/hip_bf16.h>

#define HID   128
#define NN    50000
#define NE    800000
#define TILE  32

typedef __attribute__((ext_vector_type(16))) unsigned short v16u16;
typedef __attribute__((ext_vector_type(16))) __bf16        v16bf;
typedef __attribute__((ext_vector_type(8)))  float         v8f;

__device__ __forceinline__ unsigned short f2bfu(float f) {
    union { float f; unsigned u; } v; v.f = f;
    unsigned r = v.u + 0x7FFFu + ((v.u >> 16) & 1u);   // round-to-nearest-even
    return (unsigned short)(r >> 16);
}

__device__ __forceinline__ float silu_f(float x) {
    return x / (1.0f + __expf(-x));
}

// A fragment (16x32 bf16): lane holds row M=lane%16; elems i -> K=k0+half*16+i.
// One contiguous 32B run.
__device__ __forceinline__ v16bf frag_a(const unsigned short* rowbase, int k0, int hl) {
    v16u16 t;
    const unsigned short* p0 = rowbase + k0 + hl * 16;
    const unsigned short* p1 = p0 + 8;
#pragma unroll
    for (int i = 0; i < 8; ++i) { t[i] = p0[i]; t[i + 8] = p1[i]; }
    return __builtin_bit_cast(v16bf, t);
}

// B fragment (32x16 bf16): lane holds col N=lane%16; elems i -> K=k0+half*16+i.
// Weights stored transposed [N][K] so this is one contiguous 32B run.
__device__ __forceinline__ v16bf frag_b(const unsigned short* colbase, int k0, int hl) {
    v16u16 t;
    const unsigned short* p = colbase + k0 + hl * 16;
#pragma unroll
    for (int i = 0; i < 16; ++i) t[i] = p[i];
    return __builtin_bit_cast(v16bf, t);
}

#define WMMA_BF16(A, B, C) \
    __builtin_amdgcn_wmma_f32_16x16x32_bf16(false, (A), false, (B), (short)0, (C), false, false)

// ---------------------------------------------------------------------------
// Edge pipeline: msg = silu(silu([n_s|n_r|radial]@ew0+eb0)@ew1+eb1)
//   agg[recv]    += msg                       (global f32 atomics, L2-resident)
//   w            = silu(msg@pw0+pb0)@pw1      (per-edge scalar, LDS reduce)
//   posagg[send] += clip(coord_diff*w)
// ---------------------------------------------------------------------------
__global__ __launch_bounds__(256) void egnn_edge_kernel(
    const float* __restrict__ nodes, const float* __restrict__ pos,
    const float* __restrict__ ew0, const float* __restrict__ eb0,
    const float* __restrict__ ew1, const float* __restrict__ eb1,
    const float* __restrict__ pw0, const float* __restrict__ pb0,
    const float* __restrict__ pw1,
    const int* __restrict__ senders, const int* __restrict__ receivers,
    float* __restrict__ agg, float* __restrict__ posagg)
{
    __shared__ unsigned short ew0sT[128][256];  // 64 KB, [col][k], k<256
    __shared__ unsigned short ew1T[128][128];   // 32 KB
    __shared__ unsigned short pw0T[128][128];   // 32 KB
    __shared__ float ew0last[128];              // radial row (k=256), kept f32
    __shared__ float eb0s[128], eb1s[128], pb0s[128], pw1s[128];
    __shared__ unsigned short msgin[TILE][256]; // 16 KB  bf16 activations
    __shared__ unsigned short hbuf[TILE][128];  //  8 KB
    __shared__ unsigned short msgbuf[TILE][128];//  8 KB
    __shared__ float sh_cd[TILE][3];
    __shared__ float sh_rad[TILE];
    __shared__ float sh_w[TILE];
    __shared__ int   sh_s[TILE], sh_r[TILE];

    const int tid = threadIdx.x;

    // One-time cooperative weight load (amortized over many edge tiles).
    for (int idx = tid; idx < 128 * 256; idx += 256) {
        int c = idx & 127, k = idx >> 7;
        ew0sT[c][k] = f2bfu(ew0[k * 128 + c]);
        if (k < 128) {
            ew1T[c][k] = f2bfu(ew1[k * 128 + c]);
            pw0T[c][k] = f2bfu(pw0[k * 128 + c]);
        }
    }
    if (tid < 128) {
        ew0last[tid] = ew0[256 * 128 + tid];
        eb0s[tid] = eb0[tid]; eb1s[tid] = eb1[tid];
        pb0s[tid] = pb0[tid]; pw1s[tid] = pw1[tid];
    }
    __syncthreads();

    const int lane = tid & 31;
    const int wv   = tid >> 5;     // wave id == output column tile (8 x 16 = 128 cols)
    const int hl   = lane >> 4;
    const int nsel = lane & 15;
    const int col  = wv * 16 + nsel;

    for (int tile = blockIdx.x; tile < NE / TILE; tile += gridDim.x) {
        const int e   = tid & 31;
        const int sub = tid >> 5;
        const size_t ebase = (size_t)tile * TILE;
        const int s = senders[ebase + e];
        const int r = receivers[ebase + e];

        if (sub == 0) {
            sh_s[e] = s; sh_r[e] = r;
            float dx = pos[s * 3 + 0] - pos[r * 3 + 0];
            float dy = pos[s * 3 + 1] - pos[r * 3 + 1];
            float dz = pos[s * 3 + 2] - pos[r * 3 + 2];
            sh_cd[e][0] = dx; sh_cd[e][1] = dy; sh_cd[e][2] = dz;
            sh_rad[e] = dx * dx + dy * dy + dz * dz;
            sh_w[e] = 0.0f;
        }
        // Gather [n_send | n_recv] -> bf16 LDS (float4 loads, packed b64 stores)
#pragma unroll
        for (int j = 0; j < 8; ++j) {
            int f4 = sub * 8 + j;
            const float4 v = (f4 < 32)
                ? *((const float4*)(nodes + (size_t)s * HID) + f4)
                : *((const float4*)(nodes + (size_t)r * HID) + (f4 - 32));
            unsigned long long pk =
                (unsigned long long)f2bfu(v.x)
              | ((unsigned long long)f2bfu(v.y) << 16)
              | ((unsigned long long)f2bfu(v.z) << 32)
              | ((unsigned long long)f2bfu(v.w) << 48);
            *(unsigned long long*)&msgin[e][f4 * 4] = pk;
        }
        __syncthreads();

        // ---- GEMM1: h = silu(msgin @ ew0 + eb0 + radial*ew0[256,:]) , K=256
        {
            v8f acc0 = {0,0,0,0,0,0,0,0}, acc1 = {0,0,0,0,0,0,0,0};
#pragma unroll
            for (int k0 = 0; k0 < 256; k0 += 32) {
                v16bf b  = frag_b(&ew0sT[col][0], k0, hl);
                v16bf a0 = frag_a(&msgin[nsel][0], k0, hl);
                v16bf a1 = frag_a(&msgin[16 + nsel][0], k0, hl);
                acc0 = WMMA_BF16(a0, b, acc0);
                acc1 = WMMA_BF16(a1, b, acc1);
            }
            const float bias = eb0s[col], wl = ew0last[col];
#pragma unroll
            for (int rr = 0; rr < 8; ++rr) {
                int m = rr + hl * 8;
                float v0 = silu_f(acc0[rr] + bias + sh_rad[m] * wl);
                float v1 = silu_f(acc1[rr] + bias + sh_rad[16 + m] * wl);
                hbuf[m][col]      = f2bfu(v0);
                hbuf[16 + m][col] = f2bfu(v1);
            }
        }
        __syncthreads();

        // ---- GEMM2: msg = silu(h @ ew1 + eb1) ; scatter into agg[receivers]
        {
            v8f acc0 = {0,0,0,0,0,0,0,0}, acc1 = {0,0,0,0,0,0,0,0};
#pragma unroll
            for (int k0 = 0; k0 < 128; k0 += 32) {
                v16bf b  = frag_b(&ew1T[col][0], k0, hl);
                v16bf a0 = frag_a(&hbuf[nsel][0], k0, hl);
                v16bf a1 = frag_a(&hbuf[16 + nsel][0], k0, hl);
                acc0 = WMMA_BF16(a0, b, acc0);
                acc1 = WMMA_BF16(a1, b, acc1);
            }
            const float bias = eb1s[col];
#pragma unroll
            for (int rr = 0; rr < 8; ++rr) {
                int m = rr + hl * 8;
                float v0 = silu_f(acc0[rr] + bias);
                float v1 = silu_f(acc1[rr] + bias);
                msgbuf[m][col]      = f2bfu(v0);
                msgbuf[16 + m][col] = f2bfu(v1);
                atomicAdd(&agg[(size_t)sh_r[m] * HID + col], v0);
                atomicAdd(&agg[(size_t)sh_r[16 + m] * HID + col], v1);
            }
        }
        __syncthreads();

        // ---- GEMM3: u = silu(msg @ pw0 + pb0);  w_e = sum_col u*pw1[col]
        {
            v8f acc0 = {0,0,0,0,0,0,0,0}, acc1 = {0,0,0,0,0,0,0,0};
#pragma unroll
            for (int k0 = 0; k0 < 128; k0 += 32) {
                v16bf b  = frag_b(&pw0T[col][0], k0, hl);
                v16bf a0 = frag_a(&msgbuf[nsel][0], k0, hl);
                v16bf a1 = frag_a(&msgbuf[16 + nsel][0], k0, hl);
                acc0 = WMMA_BF16(a0, b, acc0);
                acc1 = WMMA_BF16(a1, b, acc1);
            }
            const float bias = pb0s[col], pw = pw1s[col];
#pragma unroll
            for (int rr = 0; rr < 8; ++rr) {
                int m = rr + hl * 8;
                atomicAdd(&sh_w[m],      silu_f(acc0[rr] + bias) * pw);
                atomicAdd(&sh_w[16 + m], silu_f(acc1[rr] + bias) * pw);
            }
        }
        __syncthreads();

        if (tid < TILE) {
            float ww = sh_w[tid];
            int   sv = sh_s[tid];
#pragma unroll
            for (int d = 0; d < 3; ++d) {
                float tr = sh_cd[tid][d] * ww;
                tr = fminf(fmaxf(tr, -100.0f), 100.0f);
                atomicAdd(&posagg[(size_t)sv * 3 + d], tr);
            }
        }
        __syncthreads();
    }
}

// ---------------------------------------------------------------------------
// Node MLP: out = nodes + silu([nodes|agg]@nw0+nb0)@nw1 + nb1   (contiguous)
// ---------------------------------------------------------------------------
__global__ __launch_bounds__(256) void egnn_node_kernel(
    const float* __restrict__ nodes, const float* __restrict__ agg,
    const float* __restrict__ nw0, const float* __restrict__ nb0,
    const float* __restrict__ nw1, const float* __restrict__ nb1,
    float* __restrict__ out)
{
    __shared__ unsigned short nw0T[128][256];   // 64 KB
    __shared__ unsigned short nw1T[128][128];   // 32 KB
    __shared__ float nb0s[128], nb1s[128];
    __shared__ unsigned short xbuf[TILE][256];  // 16 KB
    __shared__ unsigned short hxbuf[TILE][128]; //  8 KB

    const int tid = threadIdx.x;
    for (int idx = tid; idx < 128 * 256; idx += 256) {
        int c = idx & 127, k = idx >> 7;
        nw0T[c][k] = f2bfu(nw0[k * 128 + c]);
        if (k < 128) nw1T[c][k] = f2bfu(nw1[k * 128 + c]);
    }
    if (tid < 128) { nb0s[tid] = nb0[tid]; nb1s[tid] = nb1[tid]; }
    __syncthreads();

    const int lane = tid & 31;
    const int wv   = tid >> 5;
    const int hl   = lane >> 4;
    const int nsel = lane & 15;
    const int col  = wv * 16 + nsel;
    const int NT   = (NN + TILE - 1) / TILE;

    for (int tile = blockIdx.x; tile < NT; tile += gridDim.x) {
        const int e   = tid & 31;
        const int sub = tid >> 5;
        int n  = tile * TILE + e;
        int nn = n < NN ? n : NN - 1;
#pragma unroll
        for (int j = 0; j < 8; ++j) {
            int f4 = sub * 8 + j;
            const float4 v = (f4 < 32)
                ? *((const float4*)(nodes + (size_t)nn * HID) + f4)
                : *((const float4*)(agg   + (size_t)nn * HID) + (f4 - 32));
            unsigned long long pk =
                (unsigned long long)f2bfu(v.x)
              | ((unsigned long long)f2bfu(v.y) << 16)
              | ((unsigned long long)f2bfu(v.z) << 32)
              | ((unsigned long long)f2bfu(v.w) << 48);
            *(unsigned long long*)&xbuf[e][f4 * 4] = pk;
        }
        __syncthreads();

        // hx = silu(x @ nw0 + nb0), K=256
        {
            v8f acc0 = {0,0,0,0,0,0,0,0}, acc1 = {0,0,0,0,0,0,0,0};
#pragma unroll
            for (int k0 = 0; k0 < 256; k0 += 32) {
                v16bf b  = frag_b(&nw0T[col][0], k0, hl);
                v16bf a0 = frag_a(&xbuf[nsel][0], k0, hl);
                v16bf a1 = frag_a(&xbuf[16 + nsel][0], k0, hl);
                acc0 = WMMA_BF16(a0, b, acc0);
                acc1 = WMMA_BF16(a1, b, acc1);
            }
            const float bias = nb0s[col];
#pragma unroll
            for (int rr = 0; rr < 8; ++rr) {
                int m = rr + hl * 8;
                hxbuf[m][col]      = f2bfu(silu_f(acc0[rr] + bias));
                hxbuf[16 + m][col] = f2bfu(silu_f(acc1[rr] + bias));
            }
        }
        __syncthreads();

        // out = nodes + hx @ nw1 + nb1, K=128 (residual fused at store)
        {
            v8f acc0 = {0,0,0,0,0,0,0,0}, acc1 = {0,0,0,0,0,0,0,0};
#pragma unroll
            for (int k0 = 0; k0 < 128; k0 += 32) {
                v16bf b  = frag_b(&nw1T[col][0], k0, hl);
                v16bf a0 = frag_a(&hxbuf[nsel][0], k0, hl);
                v16bf a1 = frag_a(&hxbuf[16 + nsel][0], k0, hl);
                acc0 = WMMA_BF16(a0, b, acc0);
                acc1 = WMMA_BF16(a1, b, acc1);
            }
            const float bias = nb1s[col];
#pragma unroll
            for (int rr = 0; rr < 8; ++rr) {
                int m  = rr + hl * 8;
                int n0 = tile * TILE + m;
                int n1 = tile * TILE + 16 + m;
                if (n0 < NN) out[(size_t)n0 * HID + col] =
                    nodes[(size_t)n0 * HID + col] + acc0[rr] + bias;
                if (n1 < NN) out[(size_t)n1 * HID + col] =
                    nodes[(size_t)n1 * HID + col] + acc1[rr] + bias;
            }
        }
        __syncthreads();
    }
}

__global__ void egnn_zero_kernel(float* __restrict__ p, size_t n) {
    size_t i = (size_t)blockIdx.x * blockDim.x + threadIdx.x;
    size_t stride = (size_t)gridDim.x * blockDim.x;
    for (; i < n; i += stride) p[i] = 0.0f;
}

__global__ void egnn_pos_kernel(const float* __restrict__ pos,
                                const float* __restrict__ posagg,
                                float* __restrict__ out, int n) {
    int i = blockIdx.x * blockDim.x + threadIdx.x;
    if (i < n) out[i] = pos[i] + posagg[i];
}

extern "C" void kernel_launch(void* const* d_in, const int* in_sizes, int n_in,
                              void* d_out, int out_size, void* d_ws, size_t ws_size,
                              hipStream_t stream) {
    const float* nodes = (const float*)d_in[0];
    const float* pos   = (const float*)d_in[1];
    const float* ew0   = (const float*)d_in[2];
    const float* eb0   = (const float*)d_in[3];
    const float* ew1   = (const float*)d_in[4];
    const float* eb1   = (const float*)d_in[5];
    const float* nw0   = (const float*)d_in[6];
    const float* nb0   = (const float*)d_in[7];
    const float* nw1   = (const float*)d_in[8];
    const float* nb1   = (const float*)d_in[9];
    const float* pw0   = (const float*)d_in[10];
    const float* pb0   = (const float*)d_in[11];
    const float* pw1   = (const float*)d_in[12];
    const int* senders   = (const int*)d_in[13];
    const int* receivers = (const int*)d_in[14];

    float* out_nodes = (float*)d_out;                       // [NN, HID]
    float* out_pos   = out_nodes + (size_t)NN * HID;        // [NN, 3]

    float* agg    = (float*)d_ws;                           // [NN, HID]
    float* posagg = agg + (size_t)NN * HID;                 // [NN, 3]

    const size_t zn = (size_t)NN * HID + (size_t)NN * 3;
    egnn_zero_kernel<<<2048, 256, 0, stream>>>(agg, zn);
    egnn_edge_kernel<<<768, 256, 0, stream>>>(nodes, pos, ew0, eb0, ew1, eb1,
                                              pw0, pb0, pw1, senders, receivers,
                                              agg, posagg);
    egnn_node_kernel<<<512, 256, 0, stream>>>(nodes, agg, nw0, nb0, nw1, nb1,
                                              out_nodes);
    egnn_pos_kernel<<<(NN * 3 + 255) / 256, 256, 0, stream>>>(pos, posagg,
                                                              out_pos, NN * 3);
}
